// WindowAttention_11965778887292
// MI455X (gfx1250) — compile-verified, matrix operand order checked by reading, not run on hardware
//
#include <hip/hip_runtime.h>

typedef _Float16 hf;
typedef __attribute__((ext_vector_type(8)))  _Float16 v8h;
typedef __attribute__((ext_vector_type(16))) _Float16 v16h;
typedef __attribute__((ext_vector_type(8)))  float    v8f;

// Problem constants (Swin window attention)
constexpr int BW   = 2048;        // number of windows (batch)
constexpr int NTOK = 49;          // tokens per window
constexpr int NP   = 64;          // padded token count (4 x 16)
constexpr int CDIM = 512;
constexpr int HH   = 16;          // heads
constexpr int DH   = 32;          // head dim
constexpr int NWIN = 64;          // mask windows
constexpr int MROWS = BW * NTOK;  // 100352 (divisible by 32)
constexpr int MT32  = MROWS / 32; // 3136 32-row tiles

// ---------------------------------------------------------------------------
// Fragment loader for v_wmma_f32_16x16x32_f16 (pure f16 source -> two 16B
// loads, no VALU). 16-bit A 16x32 layout: lane half hh selects K sub-blocks:
//   elements 0..7  -> K = 8*hh + 0..7
//   elements 8..15 -> K = 16 + 8*hh + 0..7
// B (32x16) uses the symmetric packing with N in place of M.
// ---------------------------------------------------------------------------
__device__ __forceinline__ v16h frag_from_f16(const hf* row, int hh) {
  v8h lo = *(const v8h*)(row + 8 * hh);
  v8h hi = *(const v8h*)(row + 16 + 8 * hh);
  v16h f;
#pragma unroll
  for (int i = 0; i < 8; ++i) { f[i] = lo[i]; f[i + 8] = hi[i]; }
  return f;
}

__device__ __forceinline__ v8f wmma16x16x32(v16h a, v16h b, v8f c) {
  return __builtin_amdgcn_wmma_f32_16x16x32_f16(false, a, false, b,
                                                (short)0, c, false, false);
}

// ---------------------------------------------------------------------------
// Kernel 0: bulk f32 -> f16 conversion (8 elements / thread, b128 in/out).
// ---------------------------------------------------------------------------
__global__ void __launch_bounds__(256)
cvt_f32_f16(const float* __restrict__ in, hf* __restrict__ out, int n8) {
  int i = blockIdx.x * blockDim.x + threadIdx.x;
  if (i >= n8) return;
  size_t base = (size_t)i * 8;
  float4 lo = ((const float4*)(in + base))[0];
  float4 hi = ((const float4*)(in + base))[1];
  v8h o;
  o[0] = (hf)lo.x; o[1] = (hf)lo.y; o[2] = (hf)lo.z; o[3] = (hf)lo.w;
  o[4] = (hf)hi.x; o[5] = (hf)hi.y; o[6] = (hf)hi.z; o[7] = (hf)hi.w;
  *(v8h*)(out + base) = o;
}

// ---------------------------------------------------------------------------
// Kernel 1: combined relative-position-bias + window-mask table.
// comb[w][h][m][n] (n padded to 64; n>=49 gets -1e30 so softmax self-masks).
// ---------------------------------------------------------------------------
__global__ void swin_prep_comb(const float* __restrict__ bias_table,
                               const int* __restrict__ rpi,
                               const float* __restrict__ mask,
                               float* __restrict__ comb) {
  int idx = blockIdx.x * blockDim.x + threadIdx.x;
  const int total = NWIN * HH * NTOK * NP;
  if (idx >= total) return;
  int col  = idx & (NP - 1);
  int tmp  = idx >> 6;
  int mrow = tmp % NTOK;
  int t2   = tmp / NTOK;
  int h    = t2 & (HH - 1);
  int w    = t2 >> 4;
  float v;
  if (col < NTOK) {
    int r = rpi[mrow * NTOK + col];
    v = bias_table[r * HH + h] + mask[(w * NTOK + mrow) * NTOK + col];
  } else {
    v = -1.0e30f;
  }
  comb[idx] = v;
}

// ---------------------------------------------------------------------------
// Kernel 2: QKV GEMM.  Y = x @ qkv_w^T + qkv_b  (M=100352, N=1536, K=512)
// One wave computes a 32x64 tile: 16 K-steps x 8 WMMAs, all-f16 operands.
// Epilogue scatters f16 into padded per-(b,h) Q/K [64][32] and V^T [32][64].
// ---------------------------------------------------------------------------
__global__ void __launch_bounds__(256)
swin_qkv_gemm(const hf* __restrict__ xh, const hf* __restrict__ qwh,
              const float* __restrict__ qb, hf* __restrict__ qws,
              hf* __restrict__ kws, hf* __restrict__ vtws) {
  const int warp = threadIdx.x >> 5;
  const int lane = threadIdx.x & 31;
  const int hh   = lane >> 4;
  const int nl   = lane & 15;
  const int wid  = blockIdx.x * 8 + warp;
  const int mt = wid / 24;          // 3136 M tiles of 32
  const int nt = wid % 24;          // 24 N tiles of 64 (N=1536)
  if (mt >= MT32) return;

  const int m0 = mt * 32;
  const int n0 = nt * 64;

  v8f acc[2][4];
#pragma unroll
  for (int i = 0; i < 2; ++i)
#pragma unroll
    for (int j = 0; j < 4; ++j)
#pragma unroll
      for (int r = 0; r < 8; ++r) acc[i][j][r] = 0.0f;

  const hf* ar0 = xh + (size_t)(m0 + nl) * CDIM;
  const hf* ar1 = xh + (size_t)(m0 + 16 + nl) * CDIM;
  const hf* br0 = qwh + (size_t)(n0 + nl) * CDIM;
  const hf* br1 = qwh + (size_t)(n0 + 16 + nl) * CDIM;
  const hf* br2 = qwh + (size_t)(n0 + 32 + nl) * CDIM;
  const hf* br3 = qwh + (size_t)(n0 + 48 + nl) * CDIM;

#pragma unroll 2
  for (int k0 = 0; k0 < CDIM; k0 += 32) {
    __builtin_prefetch(ar0 + k0 + 128, 0, 0);   // global_prefetch_b8
    v16h a0 = frag_from_f16(ar0 + k0, hh);
    v16h a1 = frag_from_f16(ar1 + k0, hh);
    v16h b0 = frag_from_f16(br0 + k0, hh);
    v16h b1 = frag_from_f16(br1 + k0, hh);
    v16h b2 = frag_from_f16(br2 + k0, hh);
    v16h b3 = frag_from_f16(br3 + k0, hh);
    acc[0][0] = wmma16x16x32(a0, b0, acc[0][0]);
    acc[1][0] = wmma16x16x32(a1, b0, acc[1][0]);
    acc[0][1] = wmma16x16x32(a0, b1, acc[0][1]);
    acc[1][1] = wmma16x16x32(a1, b1, acc[1][1]);
    acc[0][2] = wmma16x16x32(a0, b2, acc[0][2]);
    acc[1][2] = wmma16x16x32(a1, b2, acc[1][2]);
    acc[0][3] = wmma16x16x32(a0, b3, acc[0][3]);
    acc[1][3] = wmma16x16x32(a1, b3, acc[1][3]);
  }

  // Epilogue. The 64-wide n-tile never straddles a 512 boundary ->
  // `which` is wave-uniform.
  const int which = n0 >> 9;        // 0=q 1=k 2=v
  const float qscale = 0.17677669529663689f;
#pragma unroll
  for (int i = 0; i < 2; ++i) {
#pragma unroll
    for (int r = 0; r < 8; ++r) {
      int m  = m0 + i * 16 + r + 8 * hh;
      int b_ = m / NTOK;
      int t  = m - b_ * NTOK;
#pragma unroll
      for (int j = 0; j < 4; ++j) {
        int n   = n0 + 16 * j + nl;
        int rem = n & 511;
        int h   = rem >> 5;
        int d   = rem & 31;
        float v = acc[i][j][r] + qb[n];
        size_t bh = (size_t)b_ * HH + h;
        if (which == 0) {
          qws[(bh * NP + t) * DH + d] = (hf)(v * qscale);
        } else if (which == 1) {
          kws[(bh * NP + t) * DH + d] = (hf)v;
        } else {
          vtws[(bh * DH + d) * NP + t] = (hf)v;   // transposed for PV B-frags
        }
      }
    }
  }
}

// ---------------------------------------------------------------------------
// Kernel 3: attention. One wave per (window b, head h).
// For each 16-row query tile: S = Q K^T (4 WMMAs) -> +comb -> softmax
// (shfl_xor over 16-lane N-groups) -> P through LDS into A-layout ->
// O = P V (4 WMMAs), 1/rowsum scale, store f16 in proj-GEMM row order.
// ---------------------------------------------------------------------------
__global__ void __launch_bounds__(256)
swin_attn(const hf* __restrict__ qws, const hf* __restrict__ kws,
          const hf* __restrict__ vtws, const float* __restrict__ comb,
          hf* __restrict__ ao) {
  __shared__ hf plds[8][16 * NP];   // one 16x64 P tile per wave (16 KB)

  const int warp = threadIdx.x >> 5;
  const int lane = threadIdx.x & 31;
  const int hh   = lane >> 4;
  const int nl   = lane & 15;
  const int wid  = blockIdx.x * 8 + warp;
  const int b  = wid >> 4;
  const int h  = wid & 15;
  const int w  = b & (NWIN - 1);

  const hf* qbase  = qws  + ((size_t)b * HH + h) * NP * DH;
  const hf* kbase  = kws  + ((size_t)b * HH + h) * NP * DH;
  const hf* vtbase = vtws + ((size_t)b * HH + h) * DH * NP;
  const float* cbase = comb + ((size_t)(w * HH + h)) * NTOK * NP;
  hf* pl = &plds[warp][0];

  // K and V fragments are reused by all 4 query tiles: load once.
  v16h bk[4], bv[2][2];
#pragma unroll
  for (int j = 0; j < 4; ++j)
    bk[j] = frag_from_f16(kbase + (16 * j + nl) * DH, hh);
#pragma unroll
  for (int j = 0; j < 2; ++j) {
    bv[0][j] = frag_from_f16(vtbase + (16 * j + nl) * NP + 0,  hh);
    bv[1][j] = frag_from_f16(vtbase + (16 * j + nl) * NP + 32, hh);
  }

  for (int mt = 0; mt < 4; ++mt) {
    // ---- S = Q K^T over one 16x64 block -------------------------------
    v16h aq = frag_from_f16(qbase + (mt * 16 + nl) * DH, hh);
    v8f s[4];
#pragma unroll
    for (int j = 0; j < 4; ++j) {
#pragma unroll
      for (int r = 0; r < 8; ++r) s[j][r] = 0.0f;
      s[j] = wmma16x16x32(aq, bk[j], s[j]);
    }

    // ---- add fused bias+mask (cols>=49 are -1e30) ---------------------
#pragma unroll
    for (int r = 0; r < 8; ++r) {
      int mrow = mt * 16 + r + 8 * hh;
      int mc   = mrow < NTOK ? mrow : NTOK - 1;   // pad rows: clamp (discarded)
      const float* crow = cbase + mc * NP + nl;
#pragma unroll
      for (int j = 0; j < 4; ++j) s[j][r] += crow[16 * j];
    }

    // ---- softmax: stats across the 16 lanes of each N-group ----------
    float mx[8], sum[8], rinv[8];
#pragma unroll
    for (int r = 0; r < 8; ++r) {
      float m_ = s[0][r];
#pragma unroll
      for (int j = 1; j < 4; ++j) m_ = fmaxf(m_, s[j][r]);
#pragma unroll
      for (int x = 1; x < 16; x <<= 1) m_ = fmaxf(m_, __shfl_xor(m_, x, 32));
      mx[r] = m_;
    }
#pragma unroll
    for (int r = 0; r < 8; ++r) sum[r] = 0.0f;
#pragma unroll
    for (int j = 0; j < 4; ++j)
#pragma unroll
      for (int r = 0; r < 8; ++r) {
        float p = __expf(s[j][r] - mx[r]);
        s[j][r] = p;
        sum[r] += p;
      }
#pragma unroll
    for (int r = 0; r < 8; ++r) {
      float t_ = sum[r];
#pragma unroll
      for (int x = 1; x < 16; x <<= 1) t_ += __shfl_xor(t_, x, 32);
      rinv[r] = 1.0f / t_;
    }

    // ---- re-lay P (C-layout) into A-fragment layout via LDS ----------
#pragma unroll
    for (int j = 0; j < 4; ++j)
#pragma unroll
      for (int r = 0; r < 8; ++r)
        pl[(r + 8 * hh) * NP + 16 * j + nl] = (hf)s[j][r];
    asm volatile("s_wait_dscnt 0x0" ::: "memory");   // cross-lane DS dependency

    v16h pf0 = frag_from_f16(pl + nl * NP + 0,  hh);
    v16h pf1 = frag_from_f16(pl + nl * NP + 32, hh);

    // ---- O = P V  (K = 64 tokens = 2 steps; N = 32 dims = 2 tiles) ----
    v8f o[2];
#pragma unroll
    for (int j = 0; j < 2; ++j) {
#pragma unroll
      for (int r = 0; r < 8; ++r) o[j][r] = 0.0f;
      o[j] = wmma16x16x32(pf0, bv[0][j], o[j]);
      o[j] = wmma16x16x32(pf1, bv[1][j], o[j]);
    }

    // ---- store: ao[(b*49+t)*512 + h*32 + d], f16, skip pad rows -------
#pragma unroll
    for (int r = 0; r < 8; ++r) {
      int t = mt * 16 + r + 8 * hh;
      if (t < NTOK) {
        hf* orow = ao + ((size_t)b * NTOK + t) * CDIM + h * DH;
#pragma unroll
        for (int j = 0; j < 2; ++j)
          orow[16 * j + nl] = (hf)(o[j][r] * rinv[r]);
      }
    }
  }
}

// ---------------------------------------------------------------------------
// Kernel 4: output projection. out = ao @ proj_w^T + proj_b (fp32 out).
// One wave computes a 32x64 tile, all-f16 operands.
// ---------------------------------------------------------------------------
__global__ void __launch_bounds__(256)
swin_proj_gemm(const hf* __restrict__ ao, const hf* __restrict__ pwh,
               const float* __restrict__ pb, float* __restrict__ out) {
  const int warp = threadIdx.x >> 5;
  const int lane = threadIdx.x & 31;
  const int hh   = lane >> 4;
  const int nl   = lane & 15;
  const int wid  = blockIdx.x * 8 + warp;
  const int mt = wid >> 3;          // 3136 M tiles of 32
  const int nt = wid & 7;           // 8 N tiles of 64 (N=512)
  if (mt >= MT32) return;

  const int m0 = mt * 32;
  const int n0 = nt * 64;

  v8f acc[2][4];
#pragma unroll
  for (int i = 0; i < 2; ++i)
#pragma unroll
    for (int j = 0; j < 4; ++j)
#pragma unroll
      for (int r = 0; r < 8; ++r) acc[i][j][r] = 0.0f;

  const hf* ar0 = ao + (size_t)(m0 + nl) * CDIM;
  const hf* ar1 = ao + (size_t)(m0 + 16 + nl) * CDIM;
  const hf* br0 = pwh + (size_t)(n0 + nl) * CDIM;
  const hf* br1 = pwh + (size_t)(n0 + 16 + nl) * CDIM;
  const hf* br2 = pwh + (size_t)(n0 + 32 + nl) * CDIM;
  const hf* br3 = pwh + (size_t)(n0 + 48 + nl) * CDIM;

#pragma unroll 2
  for (int k0 = 0; k0 < CDIM; k0 += 32) {
    __builtin_prefetch(ar0 + k0 + 128, 0, 0);
    v16h a0 = frag_from_f16(ar0 + k0, hh);
    v16h a1 = frag_from_f16(ar1 + k0, hh);
    v16h b0 = frag_from_f16(br0 + k0, hh);
    v16h b1 = frag_from_f16(br1 + k0, hh);
    v16h b2 = frag_from_f16(br2 + k0, hh);
    v16h b3 = frag_from_f16(br3 + k0, hh);
    acc[0][0] = wmma16x16x32(a0, b0, acc[0][0]);
    acc[1][0] = wmma16x16x32(a1, b0, acc[1][0]);
    acc[0][1] = wmma16x16x32(a0, b1, acc[0][1]);
    acc[1][1] = wmma16x16x32(a1, b1, acc[1][1]);
    acc[0][2] = wmma16x16x32(a0, b2, acc[0][2]);
    acc[1][2] = wmma16x16x32(a1, b2, acc[1][2]);
    acc[0][3] = wmma16x16x32(a0, b3, acc[0][3]);
    acc[1][3] = wmma16x16x32(a1, b3, acc[1][3]);
  }

#pragma unroll
  for (int i = 0; i < 2; ++i)
#pragma unroll
    for (int j = 0; j < 4; ++j) {
      int n = n0 + 16 * j + nl;
      float bias = pb[n];
#pragma unroll
      for (int r = 0; r < 8; ++r) {
        int m = m0 + i * 16 + r + 8 * hh;
        out[(size_t)m * CDIM + n] = acc[i][j][r] + bias;
      }
    }
}

// ---------------------------------------------------------------------------
extern "C" void kernel_launch(void* const* d_in, const int* in_sizes, int n_in,
                              void* d_out, int out_size, void* d_ws,
                              size_t ws_size, hipStream_t stream) {
  const float* x      = (const float*)d_in[0];
  const float* mask   = (const float*)d_in[1];
  const int*   rpi    = (const int*)d_in[2];
  const float* qkv_w  = (const float*)d_in[3];
  const float* qkv_b  = (const float*)d_in[4];
  const float* proj_w = (const float*)d_in[5];
  const float* proj_b = (const float*)d_in[6];
  const float* btab   = (const float*)d_in[7];
  float* out = (float*)d_out;

  // Workspace layout (bytes, all 256B-aligned)
  char* ws = (char*)d_ws;
  const size_t SZ_QK = (size_t)BW * HH * NP * DH * sizeof(hf);   // 134 MB each
  const size_t SZ_AO = (size_t)MROWS * CDIM * sizeof(hf);        // 103 MB
  const size_t SZ_CB = (size_t)NWIN * HH * NTOK * NP * sizeof(float);
  size_t off = 0;
  hf*    qws  = (hf*)(ws + off); off += SZ_QK;
  hf*    kws  = (hf*)(ws + off); off += SZ_QK;
  hf*    vtws = (hf*)(ws + off); off += SZ_QK;
  hf*    ao   = (hf*)(ws + off); off += SZ_AO;
  float* comb = (float*)(ws + off); off += SZ_CB;
  hf*    xh   = (hf*)(ws + off); off += (size_t)MROWS * CDIM * sizeof(hf);
  hf*    qwh  = (hf*)(ws + off); off += (size_t)3 * CDIM * CDIM * sizeof(hf);
  hf*    pwh  = (hf*)(ws + off); off += (size_t)CDIM * CDIM * sizeof(hf);

  // 0) one-shot f32 -> f16 conversions (bandwidth-bound)
  {
    int n8 = MROWS * CDIM / 8;
    cvt_f32_f16<<<(n8 + 255) / 256, 256, 0, stream>>>(x, xh, n8);
    n8 = 3 * CDIM * CDIM / 8;
    cvt_f32_f16<<<(n8 + 255) / 256, 256, 0, stream>>>(qkv_w, qwh, n8);
    n8 = CDIM * CDIM / 8;
    cvt_f32_f16<<<(n8 + 255) / 256, 256, 0, stream>>>(proj_w, pwh, n8);
  }
  // 1) fused bias-gather + mask table (with -inf padding columns)
  {
    int total = NWIN * HH * NTOK * NP;
    swin_prep_comb<<<(total + 255) / 256, 256, 0, stream>>>(btab, rpi, mask,
                                                            comb);
  }
  // 2) QKV GEMM: 3136*24 waves, 8 waves/block
  {
    int blocks = (MT32 * 24 + 7) / 8;
    swin_qkv_gemm<<<blocks, 256, 0, stream>>>(xh, qwh, qkv_b, qws, kws, vtws);
  }
  // 3) attention: one wave per (b,h) -> 32768 waves
  {
    int blocks = (BW * HH) / 8;
    swin_attn<<<blocks, 256, 0, stream>>>(qws, kws, vtws, comb, ao);
  }
  // 4) projection GEMM: 3136*8 waves, 8 waves/block
  {
    int blocks = (MT32 * 8) / 8;
    swin_proj_gemm<<<blocks, 256, 0, stream>>>(ao, pwh, proj_b, out);
  }
}
